// GraphScoreNet_9156870275307
// MI455X (gfx1250) — compile-verified, matrix operand order checked by reading, not run on hardware
//
#include <hip/hip_runtime.h>
#include <hip/hip_bf16.h>
#include <cstdint>

typedef _Float16 v16h __attribute__((ext_vector_type(16)));
typedef _Float16 v8h  __attribute__((ext_vector_type(8)));
typedef float    v8f  __attribute__((ext_vector_type(8)));

static constexpr int BB   = 4;
static constexpr int NN   = 5000;
static constexpr int KK   = 20;
static constexpr int NB   = BB * NN;      // 20000 node rows (all batches)
static constexpr int NEPB = NN * KK;      // 100000 edges per batch
static constexpr int TPB  = NEPB / 16;    // 6250 edge tiles per batch
static constexpr int GEMM_GRID = NB / 16; // 1250

// packed weight: [8 n-tiles][4 k-chunks][32 lanes][16 halfs] = 16384 halfs = 32KB
static constexpr int PKW = 16384;

// ---------------- workspace layout (bytes) ----------------
static constexpr size_t OFF_H32  = 0;                             // f32 [NB,128]
static constexpr size_t OFF_H16  = OFF_H32  + (size_t)NB*128*4;   // f16 [NB,128]
static constexpr size_t OFF_XA   = OFF_H16  + (size_t)NB*128*2;   // f16 [NB,128]
static constexpr size_t OFF_XB   = OFF_XA   + (size_t)NB*128*2;   // f16 [NB,128]
static constexpr size_t OFF_AS   = OFF_XB   + (size_t)NB*128*2;   // f32 [NB,128] (also t1)
static constexpr size_t OFF_BR   = OFF_AS   + (size_t)NB*128*4;   // f32 [NB,128] (also agg16)
static constexpr size_t OFF_AGG  = OFF_BR   + (size_t)NB*128*4;   // f32 [NB,128] (also upd32)
static constexpr size_t OFF_RECV = OFF_AGG  + (size_t)NB*128*4;   // int [B,N,K]
static constexpr size_t OFF_SM   = OFF_RECV + (size_t)BB*NN*KK*4; // small f32 region
static constexpr size_t OFF_PACK = OFF_SM   + 32768;              // 46 packed 128x128 f16 mats

// branch-free gelu (tanh approximation): x * sigmoid(2u), u = c*(x + 0.044715 x^3)
// sigmoid via hardware v_exp_f32 / v_rcp_f32 -- no EXEC-modifying branches.
__device__ __forceinline__ float gelu_f(float x) {
    float u = 0.7978845608028654f * (x + 0.044715f * x * x * x);
    float e = __builtin_amdgcn_exp2f(-2.8853900817779268f * u);   // exp(-2u)
    return x * __builtin_amdgcn_rcpf(1.0f + e);
}

// One 16x16 output tile: A row-major [16,128] (arow = this lane's row base),
// Bp = packed weight base for THIS n-tile ([4 chunks][32 lanes][16 halfs]).
__device__ __forceinline__ v8f mm128(const _Float16* arow, const _Float16* Bp,
                                     int lane, v8f acc) {
    const int hi8 = (lane >> 4) & 1;
#pragma unroll
    for (int c = 0; c < 4; ++c) {
        const _Float16* pa = arow + c * 32 + hi8 * 8;
        v8h lo = *(const v8h*)(pa);
        v8h hi = *(const v8h*)(pa + 16);
        v16h a;
#pragma unroll
        for (int j = 0; j < 8; ++j) { a[j] = lo[j]; a[j + 8] = hi[j]; }
        v16h bv = *(const v16h*)(Bp + c * 512 + lane * 16);
        acc = __builtin_amdgcn_wmma_f32_16x16x32_f16(false, a, false, bv,
                                                     (short)0, acc, false, false);
    }
    return acc;
}

// -------- pack f32 W[row0:row0+128, 0:128] (row-major, 128 cols) into B layout --------
// out[nt][c][l][h] = W[row0 + c*32 + kloc(l,h)][nt*16 + (l&15)]
__global__ void pack_w_kernel(const float* __restrict__ W, int row0,
                              _Float16* __restrict__ out) {
    int t = blockIdx.x * blockDim.x + threadIdx.x;
    if (t >= PKW) return;
    int nt = t >> 11, c = (t >> 9) & 3, l = (t >> 4) & 31, h = t & 15;
    int kloc = (h < 8 ? h : h + 8) + ((l >= 16) ? 8 : 0);
    int n = nt * 16 + (l & 15);
    out[t] = (_Float16)W[(size_t)(row0 + c * 32 + kloc) * 128 + n];
}

// -------- cond + glob embed + per-step layer-1 "g" constants (tiny, 1 block) --------
struct CondArgs {
    const float *t, *cond;
    const float *cW0, *cb0, *cW1, *cb1, *cW2, *cb2;
    const float *gW[4], *gb[4];
    const float *We0[4], *be0[4], *Wn0[4], *bn0[4];
    float *gvec, *c1e, *c1n;   // [B,128], [4,B,128], [4,B,128]
};

__global__ void cond_kernel(CondArgs a) {
    __shared__ float cin[34], bufA[136], bufB[136], gbuf[128];
    int tid = threadIdx.x;
    for (int b = 0; b < BB; ++b) {
        if (tid < 16) {
            float fr = expf(-logf(10000.0f) * (float)tid / 15.0f);
            float ang = a.t[b] * 1000.0f * fr;
            cin[tid] = sinf(ang);
            cin[16 + tid] = cosf(ang);
        }
        if (tid < 2) cin[32 + tid] = a.cond[b * 2 + tid];
        __syncthreads();
        if (tid < 136) {
            float s = a.cb0[tid];
            for (int k = 0; k < 34; ++k) s += cin[k] * a.cW0[k * 136 + tid];
            bufA[tid] = gelu_f(s);
        }
        __syncthreads();
        if (tid < 136) {
            float s = a.cb1[tid];
            for (int k = 0; k < 136; ++k) s += bufA[k] * a.cW1[k * 136 + tid];
            bufB[tid] = gelu_f(s);
        }
        __syncthreads();
        if (tid < 34) {
            float s = a.cb2[tid];
            for (int k = 0; k < 136; ++k) s += bufB[k] * a.cW2[k * 34 + tid];
            cin[tid] = s;                      // cond output (34)
        }
        __syncthreads();
        if (tid < 128) {
            float s = a.gb[0][tid];
            for (int k = 0; k < 34; ++k) s += cin[k] * a.gW[0][k * 128 + tid];
            bufA[tid] = gelu_f(s);
        }
        __syncthreads();
        if (tid < 128) {
            float s = a.gb[1][tid];
            for (int k = 0; k < 128; ++k) s += bufA[k] * a.gW[1][k * 128 + tid];
            bufB[tid] = gelu_f(s);
        }
        __syncthreads();
        if (tid < 128) {
            float s = a.gb[2][tid];
            for (int k = 0; k < 128; ++k) s += bufB[k] * a.gW[2][k * 128 + tid];
            bufA[tid] = gelu_f(s);
        }
        __syncthreads();
        if (tid < 128) {
            float s = a.gb[3][tid];
            for (int k = 0; k < 128; ++k) s += bufA[k] * a.gW[3][k * 128 + tid];
            gbuf[tid] = s;
            a.gvec[b * 128 + tid] = s;
        }
        __syncthreads();
        if (tid < 128) {
            for (int s5 = 0; s5 < 4; ++s5) {
                float se = a.be0[s5][tid], sn = a.bn0[s5][tid];
                for (int k = 0; k < 128; ++k) {
                    float g = gbuf[k];
                    se += g * a.We0[s5][(size_t)(256 + k) * 128 + tid];
                    sn += g * a.Wn0[s5][(size_t)(256 + k) * 128 + tid];
                }
                a.c1e[s5 * (BB * 128) + b * 128 + tid] = se;
                a.c1n[s5 * (BB * 128) + b * 128 + tid] = sn;
            }
        }
        __syncthreads();
    }
}

// -------- brute-force kNN, stable-argsort-equivalent via (distbits,idx) u64 keys --------
__global__ void knn_kernel(const float* __restrict__ z, const int* __restrict__ mask,
                           int* __restrict__ recv) {
    constexpr int CAP = 5120;
    __shared__ unsigned long long keys[CAP];
    __shared__ unsigned long long red[256];
    int tid = threadIdx.x;
    int b = blockIdx.x / NN, i = blockIdx.x % NN;
    const float* zb = z + (size_t)b * NN * 3;
    float xi = zb[i * 3], yi = zb[i * 3 + 1], zi = zb[i * 3 + 2];
    bool mi = mask[b * NN + i] > 0;
    for (int j = tid; j < CAP; j += 256) {
        unsigned long long key = ~0ull;
        if (j < NN) {
            float dx = zb[j * 3] - xi, dy = zb[j * 3 + 1] - yi, dz = zb[j * 3 + 2] - zi;
            float d = dx * dx + dy * dy + dz * dz;
            if (!mi || mask[b * NN + j] <= 0) d = __builtin_inff();
            key = ((unsigned long long)__float_as_uint(d) << 32) | (unsigned)j;
        }
        keys[j] = key;
    }
    __syncthreads();
    for (int k = 0; k < KK; ++k) {
        unsigned long long m = ~0ull;
        for (int j = tid; j < CAP; j += 256) {
            unsigned long long v = keys[j];
            if (v < m) m = v;
        }
        red[tid] = m;
        __syncthreads();
        for (int s = 128; s > 0; s >>= 1) {
            if (tid < s) { unsigned long long o = red[tid + s]; if (o < red[tid]) red[tid] = o; }
            __syncthreads();
        }
        unsigned long long best = red[0];
        if (tid == 0) recv[(size_t)b * NN * KK + (size_t)i * KK + k] = (int)(best & 0xffffffffu);
        for (int j = tid; j < CAP; j += 256)
            if (keys[j] == best) keys[j] = ~0ull;
        __syncthreads();
    }
}

// -------- node_embed layer 0 (3 -> 128, gelu) --------
__global__ void embed0_kernel(const float* __restrict__ z, const float* __restrict__ W,
                              const float* __restrict__ b, _Float16* __restrict__ out) {
    int idx = blockIdx.x * 256 + threadIdx.x;
    if (idx >= NB * 128) return;
    int row = idx >> 7, f = idx & 127;
    const float* zp = z + (size_t)row * 3;
    float v = zp[0] * W[f] + zp[1] * W[128 + f] + zp[2] * W[256 + f] + b[f];
    out[idx] = (_Float16)gelu_f(v);
}

// -------- generic [NB,128]x[128,128] WMMA GEMM --------
// ACT: 0 none / 1 gelu.  BIASM: 0 none / 1 bias[128] / 2 bias[(row/NN)*128+col].
// ZADD: add f32 Zadd.  O32/O16: write f32 / f16 outputs.
template <int ACT, int BIASM, int ZADD, int O32, int O16>
__global__ void gemm128_kernel(const _Float16* __restrict__ A, const _Float16* __restrict__ Bp,
                               const float* __restrict__ bias, const float* __restrict__ Zadd,
                               float* __restrict__ Y32, _Float16* __restrict__ Y16) {
    int lane = threadIdx.x & 31, w = threadIdx.x >> 5;
    int m0 = blockIdx.x * 16, n0 = w * 16;
    v8f acc = {0.f, 0.f, 0.f, 0.f, 0.f, 0.f, 0.f, 0.f};
    acc = mm128(A + (size_t)(m0 + (lane & 15)) * 128, Bp + (size_t)w * 2048, lane, acc);
    int col = n0 + (lane & 15);
    int hi8 = (lane >> 4) & 1;
#pragma unroll
    for (int r = 0; r < 8; ++r) {
        int row = m0 + r + 8 * hi8;
        float v = acc[r];
        if (ZADD) v += Zadd[(size_t)row * 128 + col];
        if (BIASM == 1) v += bias[col];
        if (BIASM == 2) v += bias[(row / NN) * 128 + col];
        if (ACT == 1) v = gelu_f(v);
        if (O32) Y32[(size_t)row * 128 + col] = v;
        if (O16) Y16[(size_t)row * 128 + col] = (_Float16)v;
    }
}

// -------- fused edge MLP: layer1 gather -> 2x(WMMA+gelu) -> WMMA -> atomic scatter --------
__global__ void edge_mlp_kernel(const float* __restrict__ As, const float* __restrict__ Br,
                                const float* __restrict__ c1e, const int* __restrict__ recv,
                                const _Float16* __restrict__ Wp1, const float* __restrict__ b1,
                                const _Float16* __restrict__ Wp2, const float* __restrict__ b2,
                                const _Float16* __restrict__ Wp3, const float* __restrict__ b3,
                                float* __restrict__ agg) {
    __shared__ __align__(16) _Float16 xs[2][16 * 128];
    __shared__ int ridx[16];
    int tid = threadIdx.x;
    int batch = blockIdx.x / TPB;
    int e0 = (blockIdx.x % TPB) * 16;
    if (tid < 16) ridx[tid] = recv[(size_t)batch * NEPB + e0 + tid];
    __syncthreads();
    for (int idx = tid; idx < 2048; idx += 256) {
        int row = idx >> 7, col = idx & 127;
        int e = e0 + row;
        int s = e / KK;                 // sender = own node (nearest neighbor is self)
        int r = ridx[row];
        float v = As[((size_t)batch * NN + s) * 128 + col] +
                  Br[((size_t)batch * NN + r) * 128 + col] + c1e[batch * 128 + col];
        xs[0][idx] = (_Float16)gelu_f(v);
    }
    __syncthreads();
    int lane = tid & 31, w = tid >> 5;
    int col = w * 16 + (lane & 15);
    int hi8 = (lane >> 4) & 1;
    int cur = 0;
    for (int ly = 0; ly < 2; ++ly) {
        const _Float16* Wp = ((ly == 0) ? Wp1 : Wp2) + (size_t)w * 2048;
        const float* bb = (ly == 0) ? b1 : b2;
        v8f acc = {0.f, 0.f, 0.f, 0.f, 0.f, 0.f, 0.f, 0.f};
        acc = mm128(&xs[cur][(lane & 15) * 128], Wp, lane, acc);
#pragma unroll
        for (int r = 0; r < 8; ++r)
            xs[1 - cur][(r + 8 * hi8) * 128 + col] = (_Float16)gelu_f(acc[r] + bb[col]);
        __syncthreads();
        cur = 1 - cur;
    }
    v8f acc = {0.f, 0.f, 0.f, 0.f, 0.f, 0.f, 0.f, 0.f};
    acc = mm128(&xs[cur][(lane & 15) * 128], Wp3 + (size_t)w * 2048, lane, acc);
#pragma unroll
    for (int r = 0; r < 8; ++r) {
        int rr = ridx[r + 8 * hi8];
        atomicAdd(&agg[((size_t)batch * NN + rr) * 128 + col], acc[r] + b3[col]);
    }
}

__global__ void convert_kernel(const float* __restrict__ src, _Float16* __restrict__ dst) {
    int idx = blockIdx.x * 256 + threadIdx.x;
    if (idx < NB * 128) dst[idx] = (_Float16)src[idx];
}

// -------- LayerNorm(eps=1e-6) * scale + bias, skip-add into h, refresh h16 --------
__global__ void ln_skip_kernel(const float* __restrict__ upd, const float* __restrict__ scale,
                               const float* __restrict__ bias, float* __restrict__ h,
                               _Float16* __restrict__ h16) {
    __shared__ float red[128];
    int row = blockIdx.x, tid = threadIdx.x;
    float v = upd[(size_t)row * 128 + tid];
    red[tid] = v;
    __syncthreads();
    for (int s = 64; s > 0; s >>= 1) {
        if (tid < s) red[tid] += red[tid + s];
        __syncthreads();
    }
    float mu = red[0] / 128.0f;
    __syncthreads();
    float d = v - mu;
    red[tid] = d * d;
    __syncthreads();
    for (int s = 64; s > 0; s >>= 1) {
        if (tid < s) red[tid] += red[tid + s];
        __syncthreads();
    }
    float var = red[0] / 128.0f;
    float out = d * rsqrtf(var + 1e-6f) * scale[tid] + bias[tid];
    float hn = h[(size_t)row * 128 + tid] + out;
    h[(size_t)row * 128 + tid] = hn;
    h16[(size_t)row * 128 + tid] = (_Float16)hn;
}

// -------- decode last layer (128 -> 3) and eps = z - h --------
__global__ void decode_final_kernel(const _Float16* __restrict__ x16, const float* __restrict__ W,
                                    const float* __restrict__ b, const float* __restrict__ z,
                                    float* __restrict__ out) {
    int idx = blockIdx.x * 256 + threadIdx.x;
    if (idx >= NB * 3) return;
    int row = idx / 3, d = idx % 3;
    const _Float16* xp = x16 + (size_t)row * 128;
    float s = b[d];
    for (int k = 0; k < 128; ++k) s += (float)xp[k] * W[k * 3 + d];
    out[idx] = z[idx] - s;
}

extern "C" void kernel_launch(void* const* d_in, const int* in_sizes, int n_in,
                              void* d_out, int out_size, void* d_ws, size_t ws_size,
                              hipStream_t stream) {
    (void)in_sizes; (void)n_in; (void)out_size; (void)ws_size;
    auto F = [&](int i) { return (const float*)d_in[i]; };
    // leaf order: z,t,conditioning,mask, then params with sorted dict keys:
    // cond_mlp(4..9) decode(10..17) glob_embed(18..25) node_embed(26..33)
    // steps[s] at 34+18s: edge W/b x4, ln(bias,scale), node W/b x4
    const float* z = F(0);
    const float* t = F(1);
    const float* cnd = F(2);
    const int* mask = (const int*)d_in[3];

    char* ws = (char*)d_ws;
    float*    h32   = (float*)(ws + OFF_H32);
    _Float16* h16   = (_Float16*)(ws + OFF_H16);
    _Float16* xa    = (_Float16*)(ws + OFF_XA);
    _Float16* xb    = (_Float16*)(ws + OFF_XB);
    float*    As32  = (float*)(ws + OFF_AS);
    float*    Br32  = (float*)(ws + OFF_BR);
    _Float16* agg16 = (_Float16*)(ws + OFF_BR);   // alias (Br dead by then)
    float*    agg   = (float*)(ws + OFF_AGG);
    float*    upd   = (float*)(ws + OFF_AGG);     // alias (agg dead by then)
    int*      recv  = (int*)(ws + OFF_RECV);
    float*    smf   = (float*)(ws + OFF_SM);
    float*    gvec  = smf;
    float*    c1e   = smf + 512;
    float*    c1n   = smf + 512 + 2048;
    _Float16* pk    = (_Float16*)(ws + OFF_PACK);

    auto pack = [&](int slot, const float* W, int row0) {
        pack_w_kernel<<<PKW / 256, 256, 0, stream>>>(W, row0, pk + (size_t)slot * PKW);
    };
    // node_embed layers 1..3 -> slots 0..2
    pack(0, F(28), 0); pack(1, F(30), 0); pack(2, F(32), 0);
    for (int s = 0; s < 4; ++s) {
        int SB = 34 + 18 * s, ps = 3 + 10 * s;
        pack(ps + 0, F(SB + 0), 0);    // edge W0 rows 0..127   (h[sender])
        pack(ps + 1, F(SB + 0), 128);  // edge W0 rows 128..255 (h[receiver])
        pack(ps + 2, F(SB + 2), 0);    // edge W1
        pack(ps + 3, F(SB + 4), 0);    // edge W2
        pack(ps + 4, F(SB + 6), 0);    // edge W3
        pack(ps + 5, F(SB + 10), 0);   // node W0 rows 0..127   (h)
        pack(ps + 6, F(SB + 10), 128); // node W0 rows 128..255 (agg)
        pack(ps + 7, F(SB + 12), 0);   // node W1
        pack(ps + 8, F(SB + 14), 0);   // node W2
        pack(ps + 9, F(SB + 16), 0);   // node W3
    }
    pack(43, F(10), 0); pack(44, F(12), 0); pack(45, F(14), 0);  // decode 0..2

    CondArgs ca;
    ca.t = t; ca.cond = cnd;
    ca.cW0 = F(4); ca.cb0 = F(5); ca.cW1 = F(6); ca.cb1 = F(7); ca.cW2 = F(8); ca.cb2 = F(9);
    for (int i = 0; i < 4; ++i) { ca.gW[i] = F(18 + 2 * i); ca.gb[i] = F(19 + 2 * i); }
    for (int s = 0; s < 4; ++s) {
        int SB = 34 + 18 * s;
        ca.We0[s] = F(SB); ca.be0[s] = F(SB + 1);
        ca.Wn0[s] = F(SB + 10); ca.bn0[s] = F(SB + 11);
    }
    ca.gvec = gvec; ca.c1e = c1e; ca.c1n = c1n;
    cond_kernel<<<1, 256, 0, stream>>>(ca);

    knn_kernel<<<NB, 256, 0, stream>>>(z, mask, recv);

    // node embed: 3->128 scalar, then 3 WMMA layers; final layer writes h32 + h16
    embed0_kernel<<<10000, 256, 0, stream>>>(z, F(26), F(27), xa);
    gemm128_kernel<1, 1, 0, 0, 1><<<GEMM_GRID, 256, 0, stream>>>(xa, pk + (size_t)0 * PKW, F(29), nullptr, nullptr, xb);
    gemm128_kernel<1, 1, 0, 0, 1><<<GEMM_GRID, 256, 0, stream>>>(xb, pk + (size_t)1 * PKW, F(31), nullptr, nullptr, xa);
    gemm128_kernel<0, 1, 0, 1, 1><<<GEMM_GRID, 256, 0, stream>>>(xa, pk + (size_t)2 * PKW, F(33), nullptr, h32, h16);

    for (int s = 0; s < 4; ++s) {
        int SB = 34 + 18 * s, ps = 3 + 10 * s;
        // per-node halves of edge layer 1
        gemm128_kernel<0, 0, 0, 1, 0><<<GEMM_GRID, 256, 0, stream>>>(h16, pk + (size_t)(ps + 0) * PKW, nullptr, nullptr, As32, nullptr);
        gemm128_kernel<0, 0, 0, 1, 0><<<GEMM_GRID, 256, 0, stream>>>(h16, pk + (size_t)(ps + 1) * PKW, nullptr, nullptr, Br32, nullptr);
        hipMemsetAsync(agg, 0, (size_t)NB * 128 * 4, stream);
        edge_mlp_kernel<<<BB * TPB, 256, 0, stream>>>(As32, Br32, c1e + s * 512, recv,
                                                      pk + (size_t)(ps + 2) * PKW, F(SB + 3),
                                                      pk + (size_t)(ps + 3) * PKW, F(SB + 5),
                                                      pk + (size_t)(ps + 4) * PKW, F(SB + 7), agg);
        convert_kernel<<<10000, 256, 0, stream>>>(agg, agg16);
        // node MLP: t1 = h@Wn0a ; x = gelu(t1 + agg@Wn0b + c1n[b]) ; 2x gelu ; linear -> upd
        gemm128_kernel<0, 0, 0, 1, 0><<<GEMM_GRID, 256, 0, stream>>>(h16, pk + (size_t)(ps + 5) * PKW, nullptr, nullptr, As32, nullptr);
        gemm128_kernel<1, 2, 1, 0, 1><<<GEMM_GRID, 256, 0, stream>>>(agg16, pk + (size_t)(ps + 6) * PKW, c1n + s * 512, As32, nullptr, xa);
        gemm128_kernel<1, 1, 0, 0, 1><<<GEMM_GRID, 256, 0, stream>>>(xa, pk + (size_t)(ps + 7) * PKW, F(SB + 13), nullptr, nullptr, xb);
        gemm128_kernel<1, 1, 0, 0, 1><<<GEMM_GRID, 256, 0, stream>>>(xb, pk + (size_t)(ps + 8) * PKW, F(SB + 15), nullptr, nullptr, xa);
        gemm128_kernel<0, 1, 0, 1, 0><<<GEMM_GRID, 256, 0, stream>>>(xa, pk + (size_t)(ps + 9) * PKW, F(SB + 17), nullptr, upd, nullptr);
        ln_skip_kernel<<<NB, 128, 0, stream>>>(upd, F(SB + 9), F(SB + 8), h32, h16);
    }

    // decode
    gemm128_kernel<1, 1, 0, 0, 1><<<GEMM_GRID, 256, 0, stream>>>(h16, pk + (size_t)43 * PKW, F(11), nullptr, nullptr, xa);
    gemm128_kernel<1, 1, 0, 0, 1><<<GEMM_GRID, 256, 0, stream>>>(xa, pk + (size_t)44 * PKW, F(13), nullptr, nullptr, xb);
    gemm128_kernel<1, 1, 0, 0, 1><<<GEMM_GRID, 256, 0, stream>>>(xb, pk + (size_t)45 * PKW, F(15), nullptr, nullptr, xa);
    decode_final_kernel<<<(NB * 3 + 255) / 256, 256, 0, stream>>>(xa, F(16), F(17), z, (float*)d_out);
}
